// LSTMClassifier_38182259261750
// MI455X (gfx1250) — matrix operand order repaired and checked by reading, not compiled, not run on hardware
//
#include <hip/hip_runtime.h>
#include <hip/hip_bf16.h>
#include <cstdint>
#include <cstddef>

// ---------------------------------------------------------------------------
// LSTM classifier for MI455X (gfx1250), wave32 + bf16 WMMA + async-to-LDS.
//
// 16 batch groups (M=16) x 4 hidden-slice parts. Each workgroup keeps its
// 256 gate-column bf16 weight slice resident in LDS and hoisted into VGPRs
// as WMMA B-fragments. The T=2048 recurrence exchanges 16x64 h-slices via
// L2 with per-(group,t) atomic arrive counters. x[:,t,:] tiles are staged
// one step ahead with global_load_async_to_lds_b128 (ASYNCcnt pipeline).
// ---------------------------------------------------------------------------

typedef __attribute__((ext_vector_type(16))) __bf16 v16bf;
typedef __attribute__((ext_vector_type(4)))  __bf16 v4bf;
typedef __attribute__((ext_vector_type(8)))  float  v8f;

#define BSZ    256
#define TLEN   2048
#define DIN    128
#define HDIM   256
#define NPART  4      // workgroups per batch group (hidden-unit slices)
#define MTILE  16     // batch rows per group
#define NSLICE 256    // gate columns per workgroup (4 gates x 64 units)
#define UNITS  64     // hidden units per workgroup
#define THREADS 256   // 8 waves of 32

// LDS layout (bytes):
//   Whh_s  : 256*256*2 = 131072
//   Wih_s  : 256*128*2 =  65536
//   h_s    :  16*256*2 =   8192
//   x_s    :  16*128*2 =   4096
//   gates  :  16*256*4 =  16384
//   c_s    :  16* 64*4 =   4096
//   bias   :     256*4 =   1024   (ends at 230400, 16B aligned)
//   x_stage: 2*16*128*4 = 16384   (fp32 double buffer for async loads)
#define SMEM_BYTES 246784

// --- WMMA fragment loaders (contiguous k = 16h + e order, same in A and B) -------------------

// A (16-bit, 16x32 MxK): lanes 0-15 -> K=0..15 ; lanes 16-31 -> K=16..31
__device__ __forceinline__ v16bf load_a_frag(const __bf16* base, int ldk, int lane) {
  const int m  = lane & 15;
  const int kh = (lane < 16) ? 0 : 16;
  const __bf16* p = base + m * ldk;
  v16bf a;
#pragma unroll
  for (int e = 0; e < 16; ++e) a[e] = p[kh + e];
  return a;
}

// B (16-bit, 32x16 KxN): lane n<16 holds K=0..15 of column n; lanes 16-31 K=16..31.
__device__ __forceinline__ v16bf load_b_frag(const __bf16* base, int ldk, int lane) {
  const int n  = lane & 15;
  const int kb = (lane < 16) ? 0 : 16;
  const __bf16* p = base + n * ldk + kb;
  v16bf b;
#pragma unroll
  for (int e = 0; e < 16; ++e) b[e] = p[e];
  return b;
}

__device__ __forceinline__ float sigmoidf_fast(float v) {
  return 1.0f / (1.0f + __expf(-v));
}

// Issue async copy of the 16x128 fp32 x tile for timestep t into LDS stage.
// Two 16B chunks per thread -> 2 ASYNCcnt entries per wave.
__device__ __forceinline__ void issue_x_async(const float* __restrict__ x,
                                              int g, int t, float* stage, int tid) {
#pragma unroll
  for (int h = 0; h < 2; ++h) {
    const int e = h * 1024 + tid * 4;           // tile element index
    const int m = e >> 7, k = e & 127;
    const float* gp = x + ((size_t)(g * MTILE + m) * TLEN + t) * DIN + k;
    unsigned int lds_off = (unsigned int)(uintptr_t)(stage + e);
    unsigned long long ga = (unsigned long long)(uintptr_t)gp;
    asm volatile("global_load_async_to_lds_b128 %0, %1, off"
                 :: "v"(lds_off), "v"(ga) : "memory");
  }
}

__global__ void __launch_bounds__(THREADS)
lstm_persist_kernel(const float* __restrict__ x,
                    const float* __restrict__ W_ih, const float* __restrict__ W_hh,
                    const float* __restrict__ b_ih, const float* __restrict__ b_hh,
                    const float* __restrict__ W_ff, const float* __restrict__ b_ff,
                    const float* __restrict__ W_out, const float* __restrict__ b_out,
                    float* __restrict__ out,
                    void* __restrict__ hbuf_v, unsigned int* __restrict__ cnt)
{
  extern __shared__ char smem_raw[];
  __bf16* Whh_s   = reinterpret_cast<__bf16*>(smem_raw);
  __bf16* Wih_s   = Whh_s + NSLICE * HDIM;
  __bf16* h_s     = Wih_s + NSLICE * DIN;
  __bf16* x_s     = h_s + MTILE * HDIM;
  float*  gates_s = reinterpret_cast<float*>(x_s + MTILE * DIN);
  float*  c_s     = gates_s + MTILE * NSLICE;
  float*  bias_s  = c_s + MTILE * UNITS;
  float*  x_stage = bias_s + NSLICE;            // [2][2048] fp32

  __bf16* hbuf = reinterpret_cast<__bf16*>(hbuf_v);

  const int tid  = threadIdx.x;
  const int lane = tid & 31;
  const int wv   = tid >> 5;          // 8 waves
  const int g    = blockIdx.x >> 2;   // batch group 0..15
  const int p    = blockIdx.x & 3;    // hidden slice 0..3

  // kick off the async stage for t=0 immediately (overlaps weight staging)
  issue_x_async(x, g, 0, x_stage, tid);

  // --- prologue: stage weight slice (fp32 -> bf16) into LDS -----------------
  // slice col c -> global gate row r = (c/64)*256 + p*64 + (c%64):
  // slice ordered [i(64) | f(64) | g(64) | o(64)].
  for (int i4 = tid; i4 < (NSLICE * HDIM) / 4; i4 += THREADS) {
    int i = i4 * 4;
    int c = i >> 8, k = i & 255;
    int r = ((c >> 6) << 8) + (p << 6) + (c & 63);
    float4 w = *(const float4*)&W_hh[(size_t)r * HDIM + k];
    v4bf o; o[0] = (__bf16)w.x; o[1] = (__bf16)w.y; o[2] = (__bf16)w.z; o[3] = (__bf16)w.w;
    *(v4bf*)&Whh_s[c * HDIM + k] = o;
  }
  for (int i4 = tid; i4 < (NSLICE * DIN) / 4; i4 += THREADS) {
    int i = i4 * 4;
    int c = i >> 7, k = i & 127;
    int r = ((c >> 6) << 8) + (p << 6) + (c & 63);
    float4 w = *(const float4*)&W_ih[(size_t)r * DIN + k];
    v4bf o; o[0] = (__bf16)w.x; o[1] = (__bf16)w.y; o[2] = (__bf16)w.z; o[3] = (__bf16)w.w;
    *(v4bf*)&Wih_s[c * DIN + k] = o;
  }
  for (int c = tid; c < NSLICE; c += THREADS) {
    int r = ((c >> 6) << 8) + (p << 6) + (c & 63);
    bias_s[c] = b_ih[r] + b_hh[r];
  }
  for (int i = tid; i < MTILE * HDIM; i += THREADS) h_s[i] = (__bf16)0.0f;
  for (int i = tid; i < MTILE * UNITS; i += THREADS) c_s[i] = 0.0f;
  __syncthreads();

  const int col0 = wv * 32;   // wave owns slice cols [col0, col0+32)
  const int q    = wv >> 1;   // gate type: 0=i 1=f 2=g 3=o

  // --- hoist the wave's weight B-fragments from LDS into registers ---------
  v16bf bhh[8][2];
#pragma unroll
  for (int kt = 0; kt < 8; ++kt) {
    bhh[kt][0] = load_b_frag(Whh_s + col0 * HDIM + kt * 32, HDIM, lane);
    bhh[kt][1] = load_b_frag(Whh_s + (col0 + 16) * HDIM + kt * 32, HDIM, lane);
  }
  v16bf bih[4][2];
#pragma unroll
  for (int kt = 0; kt < 4; ++kt) {
    bih[kt][0] = load_b_frag(Wih_s + col0 * DIN + kt * 32, DIN, lane);
    bih[kt][1] = load_b_frag(Wih_s + (col0 + 16) * DIN + kt * 32, DIN, lane);
  }

  const float bs0 = bias_s[col0 + (lane & 15)];
  const float bs1 = bias_s[col0 + 16 + (lane & 15)];

  // --- recurrence ----------------------------------------------------------
  for (int t = 0; t < TLEN; ++t) {
    const int par = t & 1;

    // issue next timestep's async stage, then drain this timestep's
    if (t + 1 < TLEN) {
      issue_x_async(x, g, t + 1, x_stage + ((par ^ 1) << 11), tid);
      asm volatile("s_wait_asynccnt 2" ::: "memory");
    } else {
      asm volatile("s_wait_asynccnt 0" ::: "memory");
    }

    // convert own staged fp32 elements -> bf16 x_s (same lanes that loaded them)
    {
      const float* st = x_stage + (par << 11);
#pragma unroll
      for (int h = 0; h < 2; ++h) {
        const int e = h * 1024 + tid * 4;
        v4bf o;
        o[0] = (__bf16)st[e + 0]; o[1] = (__bf16)st[e + 1];
        o[2] = (__bf16)st[e + 2]; o[3] = (__bf16)st[e + 3];
        *(v4bf*)&x_s[e] = o;
      }
    }
    __syncthreads();

    // gates[m][col] = bias + sum_k h[m][k]*Whh[col][k] + sum_k x[m][k]*Wih[col][k]
    v8f acc0, acc1;
#pragma unroll
    for (int r2 = 0; r2 < 8; ++r2) { acc0[r2] = bs0; acc1[r2] = bs1; }
#pragma unroll
    for (int kt = 0; kt < 8; ++kt) {
      v16bf a = load_a_frag(h_s + kt * 32, HDIM, lane);
      acc0 = __builtin_amdgcn_wmma_f32_16x16x32_bf16(false, a, false, bhh[kt][0],
                                                     (short)0, acc0, false, false);
      acc1 = __builtin_amdgcn_wmma_f32_16x16x32_bf16(false, a, false, bhh[kt][1],
                                                     (short)0, acc1, false, false);
    }
#pragma unroll
    for (int kt = 0; kt < 4; ++kt) {
      v16bf a = load_a_frag(x_s + kt * 32, DIN, lane);
      acc0 = __builtin_amdgcn_wmma_f32_16x16x32_bf16(false, a, false, bih[kt][0],
                                                     (short)0, acc0, false, false);
      acc1 = __builtin_amdgcn_wmma_f32_16x16x32_bf16(false, a, false, bih[kt][1],
                                                     (short)0, acc1, false, false);
    }

    // activations (per-wave single gate type), scatter to gates_s[m][col]
    {
      const int n  = lane & 15;
      const int mb = (lane < 16) ? 0 : 8;
#pragma unroll
      for (int r2 = 0; r2 < 8; ++r2) {
        float g0 = acc0[r2], g1 = acc1[r2];
        if (q == 2) { g0 = tanhf(g0); g1 = tanhf(g1); }
        else        { g0 = sigmoidf_fast(g0); g1 = sigmoidf_fast(g1); }
        gates_s[(mb + r2) * NSLICE + col0 + n]      = g0;
        gates_s[(mb + r2) * NSLICE + col0 + 16 + n] = g1;
      }
    }
    __syncthreads();

    // c/h update for this WG's 16x64 hidden slice; publish h-slice to L2
    __bf16* hb = hbuf + (size_t)par * BSZ * HDIM;
    {
      const int m  = (tid * 4) >> 6;
      const int j0 = (tid * 4) & 63;
      v4bf hv4;
#pragma unroll
      for (int e = 0; e < 4; ++e) {
        int j = j0 + e;
        float ig = gates_s[m * NSLICE + j];
        float fg = gates_s[m * NSLICE + 64 + j];
        float gg = gates_s[m * NSLICE + 128 + j];
        float og = gates_s[m * NSLICE + 192 + j];
        float cv = fg * c_s[m * UNITS + j] + ig * gg;
        c_s[m * UNITS + j] = cv;
        hv4[e] = (__bf16)(og * tanhf(cv));
      }
      *(v4bf*)&hb[(size_t)(g * MTILE + m) * HDIM + (p * UNITS + j0)] = hv4;
    }

    // group barrier: 4 WGs arrive on cnt[g][t], spin with s_sleep
    __threadfence();
    __syncthreads();
    if (tid == 0) {
      unsigned int* ca = cnt + (size_t)g * TLEN + t;
      atomicAdd(ca, 1u);
      while (__hip_atomic_load(ca, __ATOMIC_ACQUIRE, __HIP_MEMORY_SCOPE_AGENT) <
             (unsigned int)NPART)
        __builtin_amdgcn_s_sleep(2);
    }
    __syncthreads();

    // gather full h(t+1) for next step (16B vector chunks)
#pragma unroll
    for (int cchunk = 0; cchunk < 2; ++cchunk) {
      int i = (cchunk * THREADS + tid) * 8;       // 8 bf16 per chunk
      int m = i >> 8, u = i & 255;
      uint4 v = *(const uint4*)(hb + (size_t)(g * MTILE + m) * HDIM + u);
      *(uint4*)(h_s + i) = v;
    }
    __syncthreads();
  }

  // --- classifier head (one WG per group) ----------------------------------
  if (p == 0) {
    const int m  = tid >> 4;
    const int u2 = tid & 15;
    float s = b_ff[u2];
    for (int k = 0; k < HDIM; ++k)
      s += (float)h_s[m * HDIM + k] * W_ff[u2 * HDIM + k];
    gates_s[m * 16 + u2] = s;   // reuse gates_s as ff buffer
    __syncthreads();
    if (tid < MTILE * 2) {
      int mm = tid >> 1, l = tid & 1;
      float z = b_out[l];
#pragma unroll
      for (int u = 0; u < 16; ++u) z += gates_s[mm * 16 + u] * W_out[l * 16 + u];
      out[(size_t)(g * MTILE + mm) * 2 + l] = sigmoidf_fast(z);
    }
  }
}

extern "C" void kernel_launch(void* const* d_in, const int* in_sizes, int n_in,
                              void* d_out, int out_size, void* d_ws, size_t ws_size,
                              hipStream_t stream) {
  (void)in_sizes; (void)n_in; (void)out_size;
  const float* x     = (const float*)d_in[0];
  const float* W_ih  = (const float*)d_in[1];
  const float* W_hh  = (const float*)d_in[2];
  const float* b_ih  = (const float*)d_in[3];
  const float* b_hh  = (const float*)d_in[4];
  const float* W_ff  = (const float*)d_in[5];
  const float* b_ff  = (const float*)d_in[6];
  const float* W_out = (const float*)d_in[7];
  const float* b_out = (const float*)d_in[8];
  float* out = (float*)d_out;

  // workspace: [0, 256KB) double-buffered bf16 h state; then arrive counters
  const size_t HBUF_BYTES = 2ull * BSZ * HDIM * 2ull;                    // 262144
  const size_t CNT_BYTES  = 16ull * (size_t)TLEN * sizeof(unsigned int); // 131072
  if (ws_size < HBUF_BYTES + CNT_BYTES) return;
  void* hbuf = d_ws;
  unsigned int* cnt = (unsigned int*)((char*)d_ws + HBUF_BYTES);

  // counters must start at zero on every call (graph replays included)
  hipMemsetAsync(cnt, 0, CNT_BYTES, stream);

  (void)hipFuncSetAttribute((const void*)lstm_persist_kernel,
                            hipFuncAttributeMaxDynamicSharedMemorySize, SMEM_BYTES);

  dim3 grid(16 * NPART);   // 64 workgroups: 16 batch groups x 4 hidden slices
  dim3 block(THREADS);
  hipLaunchKernelGGL(lstm_persist_kernel, grid, block, SMEM_BYTES, stream,
                     x, W_ih, W_hh, b_ih, b_hh, W_ff, b_ff, W_out, b_out,
                     out, hbuf, cnt);
}
